// LYT_23733989277758
// MI455X (gfx1250) — compile-verified, matrix operand order checked by reading, not run on hardware
//
#include <hip/hip_runtime.h>
#include <hip/hip_bf16.h>
#include <stdint.h>

// ---------------------------------------------------------------------------
// CDNA5 (gfx1250) implementation of lyt_forward.
// Heavy convs / projections -> implicit GEMM on v_wmma_f32_16x16x32_f16
// (f32 accumulate), block tile 128 pixels x 32 channels, A-fragment reused
// across two B subtiles (2 WMMAs per K-chunk per wave).
// Staging is branchless (clamped addresses + post-load select) and lane-
// coalesced along M so gathers batch under a single loadcnt wait.
// Bandwidth-bound / sequential stages -> VALU kernels.
// ---------------------------------------------------------------------------

typedef __attribute__((ext_vector_type(16))) _Float16 v16h;
typedef __attribute__((ext_vector_type(8)))  float    v8f;

#define ACT_NONE     0
#define ACT_RELU     1
#define ACT_TANH     2
#define ACT_SIGMOID  3
#define ACT_SILU     4
#define ACT_SOFTPLUS 5

__device__ __forceinline__ float act_apply(float v, int act) {
  switch (act) {
    case ACT_RELU:     return v > 0.f ? v : 0.f;
    case ACT_TANH:     return tanhf(v);
    case ACT_SIGMOID:  return 1.f / (1.f + __expf(-v));
    case ACT_SILU:     return v / (1.f + __expf(-v));
    case ACT_SOFTPLUS: return (v > 20.f) ? v : log1pf(__expf(v));
    default:           return v;
  }
}

__device__ __forceinline__ int iclamp(int v, int lo, int hi) {
  return v < lo ? lo : (v > hi ? hi : v);
}

// ---------------------------------------------------------------------------
// Weight packing: W (N,K) row-major f32 -> f16 [Kpad][Npad] (K-major).
// ---------------------------------------------------------------------------
__global__ void pack_matrix_f16(const float* __restrict__ W, _Float16* __restrict__ out,
                                int N, int K, int Npad, int Kpad) {
  int i = blockIdx.x * blockDim.x + threadIdx.x;
  if (i >= Npad * Kpad) return;
  int n = i % Npad, k = i / Npad;
  float v = (n < N && k < K) ? W[(size_t)n * K + k] : 0.f;
  out[(size_t)k * Npad + n] = (_Float16)v;
}

// ---------------------------------------------------------------------------
// Fragment helpers (ISA 7.12.2 layouts).
// A 16x32 f16: lane L holds M=L%16; K-halves packed 2-per-VGPR, upper lane
// half shifts K by 8; VGPRs 4..7 add 16. C/D 16x16 f32: N=lane%16,
// M = vgpr + 8*(lane/16).
// ---------------------------------------------------------------------------
__device__ __forceinline__ void load_frag(const uint32_t* base, int row, int hi,
                                          uint32_t* u) {
#pragma unroll
  for (int vv = 0; vv < 8; ++vv) {
    int kb = ((vv & 4) ? 16 : 0) + ((vv & 3) << 1) + (hi ? 8 : 0);
    u[vv] = base[(row * 32 + kb) >> 1];
  }
}

// ---------------------------------------------------------------------------
// Implicit-GEMM conv (NCHW), KS template in {1,3}, f16 WMMA, f32 accum.
// Block: 256 thr = 8 waves; tile = 128 output pixels x 32 output channels.
// Each wave owns a 16-pixel strip and both 16-channel subtiles (A reuse).
// Optional second input tensor for channel-concat (cin >= Csplit -> in2).
// out = resid ? resid + rscale*act(conv+bias) : act(conv+bias);
// input is scaled by iscale when loaded (handles conv(y+y) etc).
// ---------------------------------------------------------------------------
template <int KS>
__global__ __launch_bounds__(256)
void conv_wmma(const float* __restrict__ in, const float* in2, int Csplit,
               const _Float16* __restrict__ wpk, const float* __restrict__ bias,
               const float* resid, float rscale, float iscale,
               float* __restrict__ out,
               int Bn, int Cin, int H, int W, int Cout, int Hout, int Wout,
               int stride, int pad, int act, int Npad)
{
  const int HWout = Hout * Wout;
  const int Mtot = Bn * HWout;
  const int Ktot = Cin * KS * KS;
  const int Kpad = (Ktot + 31) & ~31;
  __shared__ _Float16 As[128 * 32];
  __shared__ _Float16 Bs[32 * 32];
  const int tid = threadIdx.x;
  const int wave = tid >> 5, lane = tid & 31;
  const int l = lane & 15, hi = lane >> 4;
  const int m_blk = blockIdx.x * 128;
  const int n0 = blockIdx.y * 32;
  v8f acc0 = {}, acc1 = {};

  __builtin_prefetch(wpk, 0, 1);

  // K-invariant decode of this thread's staging pixel.
  // Lanes walk consecutive m -> coalesced gathers for a fixed k.
  const int mA = m_blk + (tid & 127);
  const bool mok = (mA < Mtot);
  int bA = 0, ohA = 0, owA = 0;
  if (mok) { bA = mA / HWout; int r = mA % HWout; ohA = r / Wout; owA = r % Wout; }
  const int kkb = (tid >> 7) * 16;      // this thread's 16 consecutive k-halves

  for (int k0 = 0; k0 < Kpad; k0 += 32) {
    // ---- stage A tile (128 x 32): branchless clamped gather, then cvt ----
    float vals[16];
    bool  vok[16];
#pragma unroll
    for (int t = 0; t < 16; ++t) {
      int k = k0 + kkb + t;
      int cin = k / (KS * KS); int r = k % (KS * KS);   // constant divisor
      int kh = r / KS, kw = r % KS;
      int ih = ohA * stride - pad + kh;
      int iw = owA * stride - pad + kw;
      bool valid = mok && (k < Ktot) &&
                   (ih >= 0) && (ih < H) && (iw >= 0) && (iw < W);
      int ihc = iclamp(ih, 0, H - 1);
      int iwc = iclamp(iw, 0, W - 1);
      int cinc = iclamp(cin, 0, Cin - 1);
      const float* src = (cinc < Csplit) ? in : in2;
      int c  = (cinc < Csplit) ? cinc : (cinc - Csplit);
      int Cb = (cinc < Csplit) ? Csplit : (Cin - Csplit);
      vals[t] = src[((size_t)bA * Cb + c) * H * W + (size_t)ihc * W + iwc];
      vok[t] = valid;
    }
#pragma unroll
    for (int t = 0; t < 16; ++t)
      As[(tid & 127) * 32 + kkb + t] = (_Float16)(vok[t] ? vals[t] * iscale : 0.f);

    // ---- stage B tile (32 x 32), stored [n][kk] ----
    for (int t = 0; t < 4; ++t) {
      int i = tid * 4 + t;
      int n = i >> 5, kk = i & 31;
      Bs[n * 32 + kk] = wpk[(size_t)(k0 + kk) * Npad + n0 + n];
    }
    __syncthreads();

    union { v16h h; uint32_t u[8]; } afr, bfr0, bfr1;
    const uint32_t* Au = (const uint32_t*)As;
    const uint32_t* Bu = (const uint32_t*)Bs;
    load_frag(Au, wave * 16 + l, hi, afr.u);
    load_frag(Bu, l,        hi, bfr0.u);
    load_frag(Bu, 16 + l,   hi, bfr1.u);
    acc0 = __builtin_amdgcn_wmma_f32_16x16x32_f16(false, afr.h, false, bfr0.h,
                                                  (short)0, acc0, false, false);
    acc1 = __builtin_amdgcn_wmma_f32_16x16x32_f16(false, afr.h, false, bfr1.h,
                                                  (short)0, acc1, false, false);
    __syncthreads();
  }

  // ---- epilogue: bias + act + residual, incremental NCHW scatter ----
#pragma unroll
  for (int j = 0; j < 2; ++j) {
    const v8f& acc = j ? acc1 : acc0;
    int n = n0 + j * 16 + l;
    if (n >= Cout) continue;
    float bv = bias ? bias[n] : 0.f;
    int m0 = m_blk + wave * 16 + hi * 8;
    int b = 0, oh = 0, ow = 0;
    if (m0 < Mtot) { b = m0 / HWout; int r = m0 % HWout; oh = r / Wout; ow = r % Wout; }
#pragma unroll
    for (int r = 0; r < 8; ++r) {
      int m = m0 + r;
      if (m < Mtot) {
        size_t oidx = ((size_t)b * Cout + n) * (size_t)HWout + (size_t)oh * Wout + ow;
        float v = act_apply(acc[r] + bv, act);
        if (resid) v = resid[oidx] + rscale * v;
        out[oidx] = v;
      }
      if (++ow == Wout) { ow = 0; if (++oh == Hout) { oh = 0; ++b; } }
    }
  }
}

// ---------------------------------------------------------------------------
// Strided GEMM: Y[m,n] = act( sum_k X[m*sxm + k*sxk] * Wpk[k][n] + bias[n] )
// Same WMMA engine; tile 128 x 32, 2 WMMAs per K-chunk per wave.
// ---------------------------------------------------------------------------
__global__ __launch_bounds__(256)
void gemm_wmma(const float* __restrict__ X, long long sxm, long long sxk,
               const _Float16* __restrict__ wpk, int Npad,
               const float* __restrict__ bias,
               float* __restrict__ Y, long long sym, long long syn,
               int M, int K, int N, int act)
{
  const int Kpad = (K + 31) & ~31;
  __shared__ _Float16 As[128 * 32];
  __shared__ _Float16 Bs[32 * 32];
  const int tid = threadIdx.x;
  const int wave = tid >> 5, lane = tid & 31;
  const int l = lane & 15, hi = lane >> 4;
  const long long m_blk = (long long)blockIdx.x * 128;
  const int n0 = blockIdx.y * 32;
  v8f acc0 = {}, acc1 = {};

  __builtin_prefetch(wpk, 0, 1);

  const long long mA = m_blk + (tid & 127);
  const bool mok = (mA < M);
  const long long mc = mok ? mA : 0;    // clamped: loads always in-bounds
  const int kkb = (tid >> 7) * 16;

  for (int k0 = 0; k0 < Kpad; k0 += 32) {
    float vals[16];
#pragma unroll
    for (int t = 0; t < 16; ++t) {
      int k = k0 + kkb + t;
      int kc = (k < K) ? k : (K - 1);
      vals[t] = X[mc * sxm + (long long)kc * sxk];
    }
#pragma unroll
    for (int t = 0; t < 16; ++t) {
      int k = k0 + kkb + t;
      As[(tid & 127) * 32 + kkb + t] =
          (_Float16)((mok && k < K) ? vals[t] : 0.f);
    }
    for (int t = 0; t < 4; ++t) {
      int i = tid * 4 + t;
      int n = i >> 5, kk = i & 31;
      Bs[n * 32 + kk] = wpk[(size_t)(k0 + kk) * Npad + n0 + n];
    }
    __syncthreads();
    union { v16h h; uint32_t u[8]; } afr, bfr0, bfr1;
    const uint32_t* Au = (const uint32_t*)As;
    const uint32_t* Bu = (const uint32_t*)Bs;
    load_frag(Au, wave * 16 + l, hi, afr.u);
    load_frag(Bu, l,        hi, bfr0.u);
    load_frag(Bu, 16 + l,   hi, bfr1.u);
    acc0 = __builtin_amdgcn_wmma_f32_16x16x32_f16(false, afr.h, false, bfr0.h,
                                                  (short)0, acc0, false, false);
    acc1 = __builtin_amdgcn_wmma_f32_16x16x32_f16(false, afr.h, false, bfr1.h,
                                                  (short)0, acc1, false, false);
    __syncthreads();
  }

#pragma unroll
  for (int j = 0; j < 2; ++j) {
    const v8f& acc = j ? acc1 : acc0;
    int n = n0 + j * 16 + l;
    if (n >= N) continue;
    float bv = bias ? bias[n] : 0.f;
#pragma unroll
    for (int r = 0; r < 8; ++r) {
      long long m = m_blk + wave * 16 + hi * 8 + r;
      if (m < M)
        Y[m * sym + (long long)n * syn] = act_apply(acc[r] + bv, act);
    }
  }
}

// ---------------------------------------------------------------------------
// Elementwise / small kernels
// ---------------------------------------------------------------------------
__global__ void oklab_k(const float* __restrict__ img, float* __restrict__ Lc,
                        float* __restrict__ Ac, float* __restrict__ Bc, int Bn, int HW)
{
  int i = blockIdx.x * blockDim.x + threadIdx.x;
  if (i >= Bn * HW) return;
  int b = i / HW, p = i % HW;
  const float* base = img + (size_t)b * 3 * HW;
  float r = base[p], g = base[HW + p], bl = base[2 * HW + p];
  float lm = 0.4122214708f * r + 0.5363325363f * g + 0.0514459929f * bl;
  float mm = 0.2119034982f * r + 0.6806995451f * g + 0.1073969566f * bl;
  float sm = 0.0883024619f * r + 0.2817188376f * g + 0.6299787005f * bl;
  auto cbrt3 = [](float v) {
    float sg = v < 0.f ? -1.f : 1.f;
    return sg * powf(fabsf(v) + 1e-6f, 1.f / 3.f);
  };
  float l_ = cbrt3(lm), m_ = cbrt3(mm), s_ = cbrt3(sm);
  Lc[i] = 0.2104542553f * l_ + 0.793617785f  * m_ - 0.0040720468f * s_;
  Ac[i] = 1.9779984951f * l_ - 2.428592205f  * m_ + 0.4505937099f * s_;
  Bc[i] = 0.0259040371f * l_ + 0.7827717662f * m_ - 0.808675766f  * s_;
}

// depthwise 3x3 (pad=1, stride=1) with fully strided input, NCHW output
__global__ void dwconv3x3(const float* __restrict__ in, long long sb, long long sc,
                          long long sh, long long sw,
                          const float* __restrict__ wt, const float* __restrict__ bias,
                          float* __restrict__ out, int Bn, int C, int H, int W, int act)
{
  long long tot = (long long)Bn * C * H * W;
  long long i = blockIdx.x * (long long)blockDim.x + threadIdx.x;
  if (i >= tot) return;
  int wv = (int)(i % W); long long t = i / W;
  int hv = (int)(t % H); t /= H;
  int c = (int)(t % C); int b = (int)(t / C);
  const float* wp = wt + (size_t)c * 9;
  float a = bias ? bias[c] : 0.f;
  for (int kh = 0; kh < 3; ++kh)
    for (int kw = 0; kw < 3; ++kw) {
      int ih = hv - 1 + kh, iw = wv - 1 + kw;
      if (ih >= 0 && ih < H && iw >= 0 && iw < W)
        a += wp[kh * 3 + kw] * in[b * sb + c * sc + ih * sh + iw * sw];
    }
  out[i] = act_apply(a, act);
}

// out[b,c,h,w] = (xs + xs2?)[b,c,h/s,w/s] + big?[b,c,h,w]
__global__ void upsample_add(const float* __restrict__ xs, const float* xs2,
                             const float* big, float* __restrict__ out,
                             int Bn, int C, int Hs, int Ws, int s)
{
  int Hb = Hs * s, Wb = Ws * s;
  long long tot = (long long)Bn * C * Hb * Wb;
  long long i = blockIdx.x * (long long)blockDim.x + threadIdx.x;
  if (i >= tot) return;
  int w = (int)(i % Wb); long long t = i / Wb;
  int h = (int)(t % Hb); t /= Hb;
  int c = (int)(t % C); int b = (int)(t / C);
  size_t si = ((size_t)(b * C + c) * Hs + h / s) * Ws + (w / s);
  float v = xs[si];
  if (xs2) v += xs2[si];
  if (big) v += big[i];
  out[i] = v;
}

__global__ void maxpool_k(const float* __restrict__ in, float* __restrict__ out,
                          int Bn, int C, int H, int W, int k)
{
  int Ho = H / k, Wo = W / k;
  long long tot = (long long)Bn * C * Ho * Wo;
  long long i = blockIdx.x * (long long)blockDim.x + threadIdx.x;
  if (i >= tot) return;
  int w = (int)(i % Wo); long long t = i / Wo;
  int h = (int)(t % Ho); t /= Ho;
  int c = (int)(t % C); int b = (int)(t / C);
  const float* base = in + ((size_t)(b * C + c) * H) * W;
  float m = -3.4e38f;
  for (int kh = 0; kh < k; ++kh)
    for (int kw = 0; kw < k; ++kw)
      m = fmaxf(m, base[(size_t)(h * k + kh) * W + (w * k + kw)]);
  out[i] = m;
}

__global__ void axpy_k(const float* __restrict__ a, const float* b, float sc,
                       float* out, long long n)
{
  long long i = blockIdx.x * (long long)blockDim.x + threadIdx.x;
  if (i < n) out[i] = a[i] + sc * b[i];
}

// layernorm over channel dim of NCHW tensor
__global__ void ln_channels_nchw(const float* __restrict__ x, const float* __restrict__ g,
                                 const float* __restrict__ bt, float* __restrict__ out,
                                 int Bn, int C, int HW)
{
  int i = blockIdx.x * blockDim.x + threadIdx.x;
  if (i >= Bn * HW) return;
  int b = i / HW, p = i % HW;
  const float* xp = x + (size_t)b * C * HW + p;
  float mu = 0.f;
  for (int c = 0; c < C; ++c) mu += xp[(size_t)c * HW];
  mu /= C;
  float var = 0.f;
  for (int c = 0; c < C; ++c) { float d = xp[(size_t)c * HW] - mu; var += d * d; }
  var /= C;
  float inv = rsqrtf(var + 1e-5f);
  for (int c = 0; c < C; ++c)
    out[(size_t)b * C * HW + (size_t)c * HW + p] = (xp[(size_t)c * HW] - mu) * inv * g[c] + bt[c];
}

// mean over H*W per (b,c): one block per (b,c)
__global__ void mean_hw(const float* __restrict__ x, float* __restrict__ out, int HW)
{
  __shared__ float red[256];
  int bc = blockIdx.x;
  const float* xp = x + (size_t)bc * HW;
  float s = 0.f;
  for (int i = threadIdx.x; i < HW; i += 256) s += xp[i];
  red[threadIdx.x] = s;
  __syncthreads();
  for (int off = 128; off; off >>= 1) {
    if (threadIdx.x < off) red[threadIdx.x] += red[threadIdx.x + off];
    __syncthreads();
  }
  if (threadIdx.x == 0) out[bc] = red[0] / HW;
}

// tiny squeeze-excite MLP: yscale[b,co] = tanh( relu(y@fc1^T+b1) @ fc2^T + b2 )
__global__ void msef_fc(const float* __restrict__ ym, const float* __restrict__ w1,
                        const float* __restrict__ b1, const float* __restrict__ w2,
                        const float* __restrict__ b2, float* __restrict__ ys,
                        int Bn, int C, int Hd)
{
  int i = blockIdx.x * blockDim.x + threadIdx.x;
  if (i >= Bn * C) return;
  int b = i / C, co = i % C;
  float acc = b2[co];
  for (int j = 0; j < Hd; ++j) {
    float h = b1[j];
    for (int c = 0; c < C; ++c) h += w1[j * C + c] * ym[b * C + c];
    h = h > 0.f ? h : 0.f;
    acc += w2[co * Hd + j] * h;
  }
  ys[i] = tanhf(acc);
}

__global__ void msef_out_k(const float* __restrict__ x1dw, const float* __restrict__ xn,
                           const float* __restrict__ ys, const float* __restrict__ xin,
                           float* __restrict__ out, int Bn, int C, int HW)
{
  long long tot = (long long)Bn * C * HW;
  long long i = blockIdx.x * (long long)blockDim.x + threadIdx.x;
  if (i >= tot) return;
  long long t = i / HW;
  int c = (int)(t % C); int b = (int)(t / C);
  out[i] = x1dw[i] * (xn[i] * ys[b * C + c]) + xin[i];
}

// ---------------------------------------------------------------------------
// SS2D pieces
// ---------------------------------------------------------------------------
// xs[b,k,d,l] gather from xc (B,D,H,W) for 4 scan directions
__global__ void build_xs(const float* __restrict__ xc, float* __restrict__ xs,
                         int Bn, int D, int H, int W)
{
  int L = H * W;
  long long tot = (long long)Bn * 4 * D * L;
  long long i = blockIdx.x * (long long)blockDim.x + threadIdx.x;
  if (i >= tot) return;
  int l = (int)(i % L); long long t = i / L;
  int d = (int)(t % D); t /= D;
  int k = (int)(t % 4); int b = (int)(t / 4);
  int lsrc;
  if (k == 0) lsrc = l;
  else if (k == 1) { int w = l / H, h = l % H; lsrc = h * W + w; }
  else if (k == 2) lsrc = L - 1 - l;
  else { int l2 = L - 1 - l; int w = l2 / H, h = l2 % H; lsrc = h * W + w; }
  xs[i] = xc[((size_t)b * D + d) * L + lsrc];
}

// selective scan: one thread per (b, dir, d); 16-state recurrence in registers
__global__ void ss_scan(const float* __restrict__ xs, const float* __restrict__ dtb,
                        const float* __restrict__ xdbl, const float* __restrict__ A_log,
                        const float* __restrict__ Dp, float* __restrict__ ys,
                        int Bn, int Dd, int L, int Cdbl, int dr)
{
  int i = blockIdx.x * blockDim.x + threadIdx.x;
  if (i >= Bn * 4 * Dd) return;
  int d = i % Dd; int t = i / Dd;
  int k = t % 4; int b = t / 4;
  float Av[16];
#pragma unroll
  for (int n = 0; n < 16; ++n) Av[n] = -__expf(A_log[((size_t)k * Dd + d) * 16 + n]);
  float hst[16];
#pragma unroll
  for (int n = 0; n < 16; ++n) hst[n] = 0.f;
  size_t bk = (size_t)(b * 4 + k);
  const float* xp  = xs   + (bk * Dd + d) * L;
  const float* dtp = dtb  + (bk * Dd + d) * L;
  const float* Bp  = xdbl + (bk * Cdbl + dr) * L;
  const float* Cp  = xdbl + (bk * Cdbl + dr + 16) * L;
  float Dv = Dp[k * Dd + d];
  float* yp = ys + (bk * Dd + d) * L;
  for (int l = 0; l < L; ++l) {
    float dtv = dtp[l], xv = xp[l];
    float dx = dtv * xv, y = 0.f;
#pragma unroll
    for (int n = 0; n < 16; ++n) {
      hst[n] = hst[n] * __expf(dtv * Av[n]) + dx * Bp[(size_t)n * L + l];
      y += hst[n] * Cp[(size_t)n * L + l];
    }
    yp[l] = y + xv * Dv;
  }
}

// combine 4 directions + layernorm over D + SiLU gate with z (from xz)
__global__ void combine_ln_gate(const float* __restrict__ ys, const float* __restrict__ xz,
                                const float* __restrict__ g, const float* __restrict__ bt,
                                float* __restrict__ out, int Bn, int Dd, int H, int W)
{
  int L = H * W;
  int i = blockIdx.x * blockDim.x + threadIdx.x;
  if (i >= Bn * L) return;
  int b = i / L, l = i % L;
  int h = l / W, w = l % W;
  int lT = w * H + h;
  float vals[64];
  float mu = 0.f;
  for (int d = 0; d < Dd; ++d) {
    float v = ys[((size_t)(b * 4 + 0) * Dd + d) * L + l]
            + ys[((size_t)(b * 4 + 1) * Dd + d) * L + lT]
            + ys[((size_t)(b * 4 + 2) * Dd + d) * L + (L - 1 - l)]
            + ys[((size_t)(b * 4 + 3) * Dd + d) * L + (L - 1 - lT)];
    vals[d] = v; mu += v;
  }
  mu /= Dd;
  float var = 0.f;
  for (int d = 0; d < Dd; ++d) { float dd = vals[d] - mu; var += dd * dd; }
  var /= Dd;
  float inv = rsqrtf(var + 1e-5f);
  int di2 = 2 * Dd;
  for (int d = 0; d < Dd; ++d) {
    float ln = (vals[d] - mu) * inv * g[d] + bt[d];
    float z = xz[((size_t)b * L + l) * di2 + Dd + d];
    out[((size_t)b * L + l) * Dd + d] = ln * (z / (1.f + __expf(-z)));
  }
}

// ---------------------------------------------------------------------------
// Host-side orchestration
// ---------------------------------------------------------------------------
struct Arena { char* p; size_t off, cap; };
static void* aalloc(Arena& A, size_t bytes) {
  size_t o = (A.off + 255) & ~(size_t)255;
  A.off = o + bytes;
  return A.p + o;
}
static float*    af(Arena& A, size_t n) { return (float*)aalloc(A, n * sizeof(float)); }
static _Float16* ah(Arena& A, size_t n) { return (_Float16*)aalloc(A, n * sizeof(_Float16)); }

static _Float16* packW(hipStream_t s, Arena& A, const float* Wf, int N, int K) {
  int Npad = (N + 31) & ~31, Kpad = (K + 31) & ~31;
  _Float16* wpk = ah(A, (size_t)Kpad * Npad);
  int tot = Npad * Kpad;
  pack_matrix_f16<<<(tot + 255) / 256, 256, 0, s>>>(Wf, wpk, N, K, Npad, Kpad);
  return wpk;
}

static void conv2d_w(hipStream_t s, Arena& A,
                     const float* in, const float* in2, int Csplit,
                     const float* Wf, const float* bias,
                     const float* resid, float rscale, float iscale,
                     float* out, int Bn, int Cin, int H, int W, int Cout,
                     int ks, int stride, int pad, int act)
{
  int K = Cin * ks * ks;
  int Npad = (Cout + 31) & ~31;
  _Float16* wpk = packW(s, A, Wf, Cout, K);
  int Hout = (H + 2 * pad - ks) / stride + 1;
  int Wout = (W + 2 * pad - ks) / stride + 1;
  int M = Bn * Hout * Wout;
  dim3 g((unsigned)((M + 127) / 128), (unsigned)(Npad / 32));
  if (ks == 1)
    conv_wmma<1><<<g, 256, 0, s>>>(in, in2 ? in2 : in, in2 ? Csplit : Cin,
                                   wpk, bias, resid, rscale, iscale, out,
                                   Bn, Cin, H, W, Cout, Hout, Wout,
                                   stride, pad, act, Npad);
  else
    conv_wmma<3><<<g, 256, 0, s>>>(in, in2 ? in2 : in, in2 ? Csplit : Cin,
                                   wpk, bias, resid, rscale, iscale, out,
                                   Bn, Cin, H, W, Cout, Hout, Wout,
                                   stride, pad, act, Npad);
}

static void gemm_w(hipStream_t s, const float* X, long long sxm, long long sxk,
                   const _Float16* wpk, const float* bias,
                   float* Y, long long sym, long long syn,
                   int M, int K, int N, int act)
{
  int Npad = (N + 31) & ~31;
  dim3 g((unsigned)((M + 127) / 128), (unsigned)(Npad / 32));
  gemm_wmma<<<g, 256, 0, s>>>(X, sxm, sxk, wpk, Npad, bias, Y, sym, syn, M, K, N, act);
}

struct SS2DP {
  const float *in_proj_w, *conv_w, *conv_b, *x_proj_w, *dt_w, *dt_b,
              *A_log, *Dp, *ln_g, *ln_b, *out_w;
  int dm, di, dr;
};
struct DenP {
  const float *w1, *b1, *w2, *b2, *w3, *b3, *w4, *b4, *rw, *rb, *ow, *ob;
  SS2DP ss;
};

static SS2DP mkss(const float* const* F, int b, int dm) {
  SS2DP s;
  s.in_proj_w = F[b + 0]; s.conv_w = F[b + 1]; s.conv_b = F[b + 2];
  s.x_proj_w  = F[b + 3]; s.dt_w   = F[b + 4]; s.dt_b   = F[b + 5];
  s.A_log     = F[b + 6]; s.Dp     = F[b + 7]; s.ln_g   = F[b + 8];
  s.ln_b      = F[b + 9]; s.out_w  = F[b + 10];
  s.dm = dm; s.di = 2 * dm; s.dr = (dm + 15) / 16;
  return s;
}
static DenP mkden(const float* const* F, int b) {
  DenP d;
  d.w1 = F[b + 0];  d.b1 = F[b + 1];  d.w2 = F[b + 2];  d.b2 = F[b + 3];
  d.w3 = F[b + 4];  d.b3 = F[b + 5];  d.w4 = F[b + 6];  d.b4 = F[b + 7];
  d.ss = mkss(F, b + 8, 16);
  d.rw = F[b + 19]; d.rb = F[b + 20]; d.ow = F[b + 21]; d.ob = F[b + 22];
  return d;
}

// x: NCHW (B, dm, H, W)  ->  out: NCHW (B, dm, H, W)
static void run_ss2d(hipStream_t s, Arena& A, const float* x, const SS2DP& P,
                     int Bn, int H, int W, float* out)
{
  const int L = H * W, dm = P.dm, di = P.di, dr = P.dr;
  const int Cdbl = dr + 32;       // dt_rank + 2*D_STATE
  const int di2 = 2 * di;

  float* xz    = af(A, (size_t)Bn * L * di2);
  float* xc    = af(A, (size_t)Bn * di * L);
  float* xs    = af(A, (size_t)Bn * 4 * di * L);
  float* xdbl  = af(A, (size_t)Bn * 4 * Cdbl * L);
  float* dtb   = af(A, (size_t)Bn * 4 * di * L);
  float* ysb   = af(A, (size_t)Bn * 4 * di * L);
  float* gated = af(A, (size_t)Bn * L * di);

  // in_proj: (B,L,dm view of NCHW) @ W^T -> xz (B,L,2di) NHWC
  _Float16* w_in = packW(s, A, P.in_proj_w, di2, dm);
  for (int b = 0; b < Bn; ++b)
    gemm_w(s, x + (size_t)b * dm * L, 1, L, w_in, nullptr,
           xz + (size_t)b * L * di2, di2, 1, L, dm, di2, ACT_NONE);

  // depthwise conv3x3 + SiLU on xin half of xz, output NCHW (B,di,L)
  {
    long long tot = (long long)Bn * di * L;
    dwconv3x3<<<(unsigned)((tot + 255) / 256), 256, 0, s>>>(
        xz, (long long)L * di2, 1LL, (long long)W * di2, (long long)di2,
        P.conv_w, P.conv_b, xc, Bn, di, H, W, ACT_SILU);
  }

  // 4-direction stack
  {
    long long tot = (long long)Bn * 4 * di * L;
    build_xs<<<(unsigned)((tot + 255) / 256), 256, 0, s>>>(xc, xs, Bn, di, H, W);
  }

  // x_proj per direction: xdbl[b,k,c,l]
  for (int k = 0; k < 4; ++k) {
    _Float16* wxp = packW(s, A, P.x_proj_w + (size_t)k * Cdbl * di, Cdbl, di);
    for (int b = 0; b < Bn; ++b)
      gemm_w(s, xs + ((size_t)(b * 4 + k) * di) * L, 1, L, wxp, nullptr,
             xdbl + ((size_t)(b * 4 + k) * Cdbl) * L, 1, L, L, di, Cdbl, ACT_NONE);
  }

  // dt projection + softplus
  for (int k = 0; k < 4; ++k) {
    _Float16* wdt = packW(s, A, P.dt_w + (size_t)k * di * dr, di, dr);
    for (int b = 0; b < Bn; ++b)
      gemm_w(s, xdbl + ((size_t)(b * 4 + k) * Cdbl) * L, 1, L, wdt, P.dt_b + k * di,
             dtb + ((size_t)(b * 4 + k) * di) * L, 1, L, L, dr, di, ACT_SOFTPLUS);
  }

  // selective scan
  {
    int tot = Bn * 4 * di;
    ss_scan<<<(tot + 63) / 64, 64, 0, s>>>(xs, dtb, xdbl, P.A_log, P.Dp, ysb,
                                           Bn, di, L, Cdbl, dr);
  }

  // combine directions + LN + SiLU gate -> gated (B,L,di)
  combine_ln_gate<<<(Bn * L + 255) / 256, 256, 0, s>>>(ysb, xz, P.ln_g, P.ln_b,
                                                       gated, Bn, di, H, W);

  // out_proj -> NCHW
  _Float16* wo = packW(s, A, P.out_w, dm, di);
  for (int b = 0; b < Bn; ++b)
    gemm_w(s, gated + (size_t)b * L * di, di, 1, wo, nullptr,
           out + (size_t)b * dm * L, 1, L, L, di, dm, ACT_NONE);
}

// x: (B,1,H,W)  ->  out = x + denoiser(x), (B,1,H,W)
static void run_denoiser(hipStream_t s, Arena& A, const float* x, const DenP& P,
                         int Bn, int H, int W, float* out)
{
  const int nf = 16;
  int H2 = H / 2, W2 = W / 2, H4 = H / 4, W4 = W / 4, H8 = H / 8, W8 = W / 8;

  float* x1 = af(A, (size_t)Bn * nf * H * W);
  conv2d_w(s, A, x, nullptr, 0, P.w1, P.b1, nullptr, 0.f, 1.f, x1,
           Bn, 1, H, W, nf, 3, 1, 1, ACT_RELU);
  float* x2 = af(A, (size_t)Bn * nf * H2 * W2);
  conv2d_w(s, A, x1, nullptr, 0, P.w2, P.b2, nullptr, 0.f, 1.f, x2,
           Bn, nf, H, W, nf, 3, 2, 1, ACT_RELU);
  float* x3 = af(A, (size_t)Bn * nf * H4 * W4);
  conv2d_w(s, A, x2, nullptr, 0, P.w3, P.b3, nullptr, 0.f, 1.f, x3,
           Bn, nf, H2, W2, nf, 3, 2, 1, ACT_RELU);
  float* x4 = af(A, (size_t)Bn * nf * H8 * W8);
  conv2d_w(s, A, x3, nullptr, 0, P.w4, P.b4, nullptr, 0.f, 1.f, x4,
           Bn, nf, H4, W4, nf, 3, 2, 1, ACT_RELU);

  float* sso = af(A, (size_t)Bn * nf * H8 * W8);
  { size_t sn = A.off; run_ss2d(s, A, x4, P.ss, Bn, H8, W8, sso); A.off = sn; }

  float* u1 = af(A, (size_t)Bn * nf * H4 * W4);
  upsample_add<<<(unsigned)(((long long)Bn * nf * H4 * W4 + 255) / 256), 256, 0, s>>>(
      sso, nullptr, nullptr, u1, Bn, nf, H8, W8, 2);
  float* u2 = af(A, (size_t)Bn * nf * H2 * W2);
  upsample_add<<<(unsigned)(((long long)Bn * nf * H2 * W2 + 255) / 256), 256, 0, s>>>(
      u1, x3, nullptr, u2, Bn, nf, H4, W4, 2);
  float* u3 = af(A, (size_t)Bn * nf * H * W);
  upsample_add<<<(unsigned)(((long long)Bn * nf * H * W + 255) / 256), 256, 0, s>>>(
      u2, x2, nullptr, u3, Bn, nf, H2, W2, 2);
  axpy_k<<<(unsigned)(((long long)Bn * nf * H * W + 255) / 256), 256, 0, s>>>(
      u3, x1, 1.f, u3, (long long)Bn * nf * H * W);

  float* yr = af(A, (size_t)Bn * H * W);
  conv2d_w(s, A, u3, nullptr, 0, P.rw, P.rb, nullptr, 0.f, 1.f, yr,
           Bn, nf, H, W, 1, 3, 1, 1, ACT_NONE);
  // out = x + tanh(conv(y + y)) : input scale 2, residual x
  conv2d_w(s, A, yr, nullptr, 0, P.ow, P.ob, x, 1.f, 2.f, out,
           Bn, 1, H, W, 1, 3, 1, 1, ACT_TANH);
}

// ---------------------------------------------------------------------------
// Entry point
// ---------------------------------------------------------------------------
extern "C" void kernel_launch(void* const* d_in, const int* in_sizes, int n_in,
                              void* d_out, int out_size, void* d_ws, size_t ws_size,
                              hipStream_t stream)
{
  (void)in_sizes; (void)n_in; (void)out_size;
  const float* const* F = (const float* const*)d_in;
  hipStream_t s = stream;
  Arena A{ (char*)d_ws, 0, ws_size };

  const int Bn = 2, H = 512, W = 512;
  const int HW = H * W;
  const int Fc = 32;

  // ---- oklab split ----
  float* lP = af(A, (size_t)Bn * HW);
  float* aP = af(A, (size_t)Bn * HW);
  float* bP = af(A, (size_t)Bn * HW);
  oklab_k<<<(Bn * HW + 255) / 256, 256, 0, s>>>(F[0], lP, aP, bP, Bn, HW);

  float* aD = af(A, (size_t)Bn * HW);
  float* bD = af(A, (size_t)Bn * HW);

  // ---- denoisers on a & b (scratch reused) ----
  DenP dA = mkden(F, 7), dB = mkden(F, 30);
  size_t snap = A.off;
  run_denoiser(s, A, aP, dA, Bn, H, W, aD);
  A.off = snap;
  run_denoiser(s, A, bP, dB, Bn, H, W, bD);
  A.off = snap;

  // ---- stem convs ----
  float* lp = af(A, (size_t)Bn * Fc * HW);
  float* ap = af(A, (size_t)Bn * Fc * HW);
  float* bp = af(A, (size_t)Bn * Fc * HW);
  conv2d_w(s, A, lP, nullptr, 0, F[1], F[2], nullptr, 0.f, 1.f, lp,
           Bn, 1, H, W, Fc, 3, 1, 1, ACT_RELU);
  conv2d_w(s, A, aD, nullptr, 0, F[3], F[4], nullptr, 0.f, 1.f, ap,
           Bn, 1, H, W, Fc, 3, 1, 1, ACT_RELU);
  conv2d_w(s, A, bD, nullptr, 0, F[5], F[6], nullptr, 0.f, 1.f, bp,
           Bn, 1, H, W, Fc, 3, 1, 1, ACT_RELU);

  // ---- luminance branch: maxpool8 -> ss2d -> up8 + lp ----
  float* lum1 = af(A, (size_t)Bn * Fc * 64 * 64);
  maxpool_k<<<(unsigned)(((long long)Bn * Fc * 64 * 64 + 255) / 256), 256, 0, s>>>(
      lp, lum1, Bn, Fc, H, W, 8);
  float* ss2 = af(A, (size_t)Bn * Fc * 4096);
  SS2DP Pss = mkss(F, 53, 32);
  { size_t sn = A.off; run_ss2d(s, A, lum1, Pss, Bn, 64, 64, ss2); A.off = sn; }
  float* lum = af(A, (size_t)Bn * Fc * HW);
  upsample_add<<<(unsigned)(((long long)Bn * Fc * HW + 255) / 256), 256, 0, s>>>(
      ss2, nullptr, lp, lum, Bn, Fc, 64, 64, 8);

  // ---- ref path: 1x1 conv over concat(ap,bp), then +0.2*conv1x1(lum) ----
  float* ref1 = af(A, (size_t)Bn * Fc * HW);
  conv2d_w(s, A, ap, bp, Fc, F[66], F[67], nullptr, 0.f, 1.f, ref1,
           Bn, 2 * Fc, H, W, Fc, 1, 1, 0, ACT_NONE);
  float* ref2 = ap;  // ap/bp/lp are dead from here on; reuse their storage
  conv2d_w(s, A, lum, nullptr, 0, F[64], F[65], ref1, 0.2f, 1.f, ref2,
           Bn, Fc, H, W, Fc, 1, 1, 0, ACT_NONE);

  // ---- MSEF ----
  float* xn = bp;
  ln_channels_nchw<<<(Bn * HW + 255) / 256, 256, 0, s>>>(ref2, F[68], F[69], xn, Bn, Fc, HW);
  float* x1dw = lp;
  dwconv3x3<<<(unsigned)(((long long)Bn * Fc * HW + 255) / 256), 256, 0, s>>>(
      xn, (long long)Fc * HW, (long long)HW, (long long)W, 1LL,
      F[70], F[71], x1dw, Bn, Fc, H, W, ACT_NONE);
  float* ymean  = af(A, (size_t)Bn * Fc);
  float* yscale = af(A, (size_t)Bn * Fc);
  mean_hw<<<Bn * Fc, 256, 0, s>>>(xn, ymean, HW);
  msef_fc<<<1, 64, 0, s>>>(ymean, F[72], F[73], F[74], F[75], yscale, Bn, Fc, 2);
  float* mtmp = af(A, (size_t)Bn * Fc * HW);
  msef_out_k<<<(unsigned)(((long long)Bn * Fc * HW + 255) / 256), 256, 0, s>>>(
      x1dw, xn, yscale, ref2, mtmp, Bn, Fc, HW);
  // ref = msef(ref) + shortcut   (write into ref1)
  axpy_k<<<(unsigned)(((long long)Bn * Fc * HW + 255) / 256), 256, 0, s>>>(
      mtmp, ref1, 1.f, ref1, (long long)Bn * Fc * HW);

  // ---- reconstruction: 3x3 conv over concat(ref, lum), then fin+sigmoid ----
  float* rec = mtmp;
  conv2d_w(s, A, ref1, lum, Fc, F[76], F[77], nullptr, 0.f, 1.f, rec,
           Bn, 2 * Fc, H, W, Fc, 3, 1, 1, ACT_NONE);
  conv2d_w(s, A, rec, nullptr, 0, F[78], F[79], nullptr, 0.f, 1.f, (float*)d_out,
           Bn, Fc, H, W, 3, 3, 1, 1, ACT_SIGMOID);
}
